// MPNNConv_42941083025512
// MI455X (gfx1250) — matrix operand order repaired and checked by reading, not compiled, not run on hardware
//
#include <hip/hip_runtime.h>
#include <hip/hip_bf16.h>

// ---------------------------------------------------------------------------
// MI455X (gfx1250) implementation of the CWN/MPNN layer.
//
// Roofline: ~79 GFLOP of GEMM vs ~550 MB HBM traffic (~24us @ 23.3 TB/s).
// fp32 WMMA (16x16x4) would be ~10x compute-bound, so all GEMMs run as bf16
// A/B with fp32 accumulation via v_wmma_f32_16x16x32_bf16 (wave32).
// B panels that are pre-converted to fragment-ordered bf16 are staged into
// LDS with global_load_async_to_lds_b128 (ASYNCcnt) and read back with
// ds_load_b128; WMMA B operands are double-buffered in registers so loads
// overlap the matrix pipe instead of serializing on s_wait.
// ---------------------------------------------------------------------------

typedef __attribute__((ext_vector_type(16))) __bf16 v16bf;
typedef __attribute__((ext_vector_type(8)))  float  v8f;

#define DD   128             // feature dim D == H
#define NN   128             // nodes per graph
#define NGR  128             // graphs B
#define EUP  (NGR*NN*16)     // 262144 messages
#define NTOT (NGR*NN)        // 16384 nodes
#define ETOT (NGR*NN*16)     // 262144 edges
#define BN_EPS 1e-5f

// Weight segment offsets inside the bf16 fragment-ordered weight pool (halves)
#define WOFF_MSG 0           // 8 ksteps  (K=256)
#define WOFF_FB  32768       // 12 ksteps (K=384)
#define WOFF_UN  81920       // 3 x 4 ksteps (K=128 each)
#define WOFF_UE  131072      // 3 x 4 ksteps

__device__ __forceinline__ v8f vzero8() {
  v8f z;
#pragma unroll
  for (int i = 0; i < 8; ++i) z[i] = 0.f;
  return z;
}

__device__ __forceinline__ v8f wmma_bf16(v16bf a, v16bf b, v8f c) {
  return __builtin_amdgcn_wmma_f32_16x16x32_bf16(false, a, false, b,
                                                 (short)0, c, false, false);
}

// A fragment (16x32 bf16) from a row of f32 data, converted in-register.
// lanes 0-15: K 0..15, lanes 16-31: K 16..31, M = lane&15.
__device__ __forceinline__ v16bf load_a_f32(const float* __restrict__ row,
                                            int k0, int hf) {
  v16bf a;
#pragma unroll
  for (int p = 0; p < 8; ++p) {
    const int e = 2 * p;
    const int k = k0 + hf * 16 + e;
    float2 f = *(const float2*)(row + k);
    a[e]     = (__bf16)f.x;
    a[e + 1] = (__bf16)f.y;
  }
  return a;
}

// B fragment f (== ks*8+nt) from a fragment-ordered bf16 pool (global or LDS).
__device__ __forceinline__ v16bf load_frag_idx(const __bf16* __restrict__ seg,
                                               int f, int lane) {
  return *(const v16bf*)(seg + (((f << 5) + lane) << 4));
}

// CDNA5 async copy: 16B global -> LDS per lane, tracked by ASYNCcnt.
__device__ __forceinline__ void async_copy16(unsigned lds_byte,
                                             const void* gsrc) {
  asm volatile("global_load_async_to_lds_b128 %0, %1, off"
               :: "v"(lds_byte), "v"(gsrc) : "memory");
}
__device__ __forceinline__ void wait_async0() {
  asm volatile("s_wait_asynccnt 0x0" ::: "memory");
}

// Stage 32KB of fragment-ordered bf16 (2048 x 16B chunks) into LDS.
__device__ __forceinline__ void stage_panel_async(__bf16* lds, int tid,
                                                  const __bf16* gsrc) {
  const unsigned ldsbase = (unsigned)(size_t)(&lds[0]);
#pragma unroll
  for (int i = 0; i < 8; ++i) {
    const int c = tid + 256 * i;
    async_copy16(ldsbase + c * 16, (const char*)gsrc + c * 16);
  }
  wait_async0();
  __syncthreads();
}

// ---------------------------------------------------------------------------
// Prep A: weights -> fragment-ordered bf16; zero BN stat accumulators.
// ---------------------------------------------------------------------------
__global__ void prep_kernel(const float* __restrict__ Wmsg,
                            const float* __restrict__ Wfb,
                            const float* __restrict__ unW,
                            const float* __restrict__ ueW,
                            __bf16* __restrict__ wsW,
                            float* __restrict__ stats) {
  const int seg = blockIdx.y;
  const float* src;
  __bf16* dst;
  int ksteps;
  switch (seg) {
    case 0: src = Wmsg; dst = wsW + WOFF_MSG; ksteps = 8;  break;
    case 1: src = Wfb;  dst = wsW + WOFF_FB;  ksteps = 12; break;
    case 2: case 3: case 4:
      src = unW + (seg - 2) * (DD * DD);
      dst = wsW + WOFF_UN + (seg - 2) * 16384; ksteps = 4; break;
    default:
      src = ueW + (seg - 5) * (DD * DD);
      dst = wsW + WOFF_UE + (seg - 5) * 16384; ksteps = 4; break;
  }
  const int total = ksteps * 4096;
  for (int t = blockIdx.x * blockDim.x + threadIdx.x; t < total;
       t += gridDim.x * blockDim.x) {
    const int e    = t & 15;
    const int lane = (t >> 4) & 31;
    const int nt   = (t >> 9) & 7;
    const int ks   = t >> 12;
    const int k = ks * 32 + ((lane & 16) ? 16 : 0) + e;
    const int n = nt * 16 + (lane & 15);
    dst[t] = (__bf16)src[k * DD + n];
  }
  if (seg == 0) {
    for (int t = blockIdx.x * blockDim.x + threadIdx.x; t < 1536;
         t += gridDim.x * blockDim.x)
      stats[t] = 0.f;
  }
}

// ---------------------------------------------------------------------------
// Prep B: n_x -> per-graph B-fragment-ordered bf16 (async-copyable panels).
// ---------------------------------------------------------------------------
__global__ void prep_nx_kernel(const float* __restrict__ n_x,
                               __bf16* __restrict__ nxf) {
  const int total = NTOT * DD;
  for (int t = blockIdx.x * blockDim.x + threadIdx.x; t < total;
       t += gridDim.x * blockDim.x) {
    const int b  = t >> 14;        // 16384 halves per graph
    const int t2 = t & 16383;
    const int e    = t2 & 15;
    const int lane = (t2 >> 4) & 31;
    const int nt   = (t2 >> 9) & 7;
    const int ks   = t2 >> 12;
    const int k = ks * 32 + ((lane & 16) ? 16 : 0) + e;   // node index
    const int n = nt * 16 + (lane & 15);                  // feature
    nxf[t] = (__bf16)n_x[((size_t)b * NN + k) * DD + n];
  }
}

// ---------------------------------------------------------------------------
// Message kernel: fused msg-MLP + gate-MLP + adjacency-weighted segment sum
// + GIN residual.  2048 blocks x 8 waves; wave w owns one node's 16 messages.
// B fragments double-buffered in registers (weights stay L1-resident; the
// 160KB msg+fb pool exceeds static LDS).
// ---------------------------------------------------------------------------
__global__ __launch_bounds__(256)
void msg_kernel(const float* __restrict__ n_x,
                const float* __restrict__ up_attr,
                const float* __restrict__ up_adj,
                const float* __restrict__ b_msg,
                const float* __restrict__ b_fb,
                const int* __restrict__ up_b,
                const int* __restrict__ up_i,
                const int* __restrict__ up_j,
                const int* __restrict__ idx_i,
                const int* __restrict__ idx_j,
                const __bf16* __restrict__ wmsg,
                const __bf16* __restrict__ wfb,
                const float* __restrict__ eps1p,
                float* __restrict__ n_pre) {
  __shared__ float lds_we[128];
  const int tid = threadIdx.x;
  const int mbase = blockIdx.x * 128;
  if (tid < 128) {
    const int m = mbase + tid;
    lds_we[tid] = up_adj[((size_t)up_b[m] * NN + up_i[m]) * NN + up_j[m]];
  }
  __syncthreads();

  const int w = tid >> 5, lane = tid & 31, hf = lane >> 4, l15 = lane & 15;
  const int ms = mbase + w * 16;
  const int rowA = ms + l15;
  const float* xi = n_x + (size_t)idx_i[rowA] * DD;
  const float* xj = n_x + (size_t)idx_j[rowA] * DD;
  const float* ua = up_attr + (size_t)rowA * DD;

  v8f accM[8], accF[8];
#pragma unroll
  for (int nt = 0; nt < 8; ++nt) { accM[nt] = vzero8(); accF[nt] = vzero8(); }

  // chunk 0: x_i -> FB rows 0..127 (frags 0..31), single stream
  {
    v16bf b0 = load_frag_idx(wfb, 0, lane);
#pragma unroll
    for (int ks = 0; ks < 4; ++ks) {
      v16bf a = load_a_f32(xi, ks * 32, hf);
#pragma unroll
      for (int nt = 0; nt < 8; ++nt) {
        const int j = ks * 8 + nt + 1;
        v16bf nb;
        if (j < 32) nb = load_frag_idx(wfb, j, lane);
        accF[nt] = wmma_bf16(a, b0, accF[nt]);
        if (j < 32) b0 = nb;
      }
    }
  }
  // chunk 1: x_j -> MSG frags 0..31, FB frags 32..63 (dual stream)
  {
    v16bf bM = load_frag_idx(wmsg, 0, lane);
    v16bf bF = load_frag_idx(wfb, 32, lane);
#pragma unroll
    for (int ks = 0; ks < 4; ++ks) {
      v16bf a = load_a_f32(xj, ks * 32, hf);
#pragma unroll
      for (int nt = 0; nt < 8; ++nt) {
        const int j = ks * 8 + nt + 1;
        v16bf nM, nF;
        if (j < 32) { nM = load_frag_idx(wmsg, j, lane);
                      nF = load_frag_idx(wfb, 32 + j, lane); }
        accM[nt] = wmma_bf16(a, bM, accM[nt]);
        accF[nt] = wmma_bf16(a, bF, accF[nt]);
        if (j < 32) { bM = nM; bF = nF; }
      }
    }
  }
  // chunk 2: up_attr -> MSG frags 32..63, FB frags 64..95
  {
    v16bf bM = load_frag_idx(wmsg, 32, lane);
    v16bf bF = load_frag_idx(wfb, 64, lane);
#pragma unroll
    for (int ks = 0; ks < 4; ++ks) {
      v16bf a = load_a_f32(ua, ks * 32, hf);
#pragma unroll
      for (int nt = 0; nt < 8; ++nt) {
        const int j = ks * 8 + nt + 1;
        v16bf nM, nF;
        if (j < 32) { nM = load_frag_idx(wmsg, 32 + j, lane);
                      nF = load_frag_idx(wfb, 64 + j, lane); }
        accM[nt] = wmma_bf16(a, bM, accM[nt]);
        accF[nt] = wmma_bf16(a, bF, accF[nt]);
        if (j < 32) { bM = nM; bF = nF; }
      }
    }
  }

  // Epilogue: bias + relu + sigmoid + w_e; reduce 16 rows (M = r + hf*8);
  // GIN residual; one node row per wave.
  const int node = up_b[ms] * NN + up_i[ms];
  const float e1 = 1.0f + eps1p[0];
#pragma unroll
  for (int nt = 0; nt < 8; ++nt) {
    const int col = nt * 16 + l15;
    const float bm = b_msg[col];
    const float bf = b_fb[col];
    float cs = 0.f;
#pragma unroll
    for (int r = 0; r < 8; ++r) {
      const float msg  = fmaxf(accM[nt][r] + bm, 0.f);
      const float gate = 1.f / (1.f + __expf(-(accF[nt][r] + bf)));
      cs += lds_we[w * 16 + hf * 8 + r] * gate * msg;
    }
    cs += __shfl_xor(cs, 16, 32);
    if (hf == 0)
      n_pre[(size_t)node * DD + col] = cs + e1 * n_x[(size_t)node * DD + col];
  }
}

// ---------------------------------------------------------------------------
// Boundary einsum: e_pre[b,e,:] = boundary_adj[b,e,:] @ n_x[b] + (1+eps2)*e_x.
// Pre-converted bf16 panel for graph b async-copied into LDS.
// ---------------------------------------------------------------------------
__global__ __launch_bounds__(256)
void boundary_kernel(const __bf16* __restrict__ nxf,
                     const float* __restrict__ e_x,
                     const float* __restrict__ badj,
                     const float* __restrict__ eps2p,
                     float* __restrict__ e_pre) {
  __shared__ __attribute__((aligned(32))) __bf16 ldsB[16384];
  const int b = blockIdx.x >> 4, rt = blockIdx.x & 15;
  const int tid = threadIdx.x;
  stage_panel_async(ldsB, tid, nxf + (size_t)b * 16384);

  const int w = tid >> 5, lane = tid & 31, hf = lane >> 4, l15 = lane & 15;
  const int row0 = rt * 128 + w * 16;
  const float* arow = badj + ((size_t)b * 2048 + row0 + l15) * NN;

  v16bf afr[4];
#pragma unroll
  for (int ks = 0; ks < 4; ++ks) afr[ks] = load_a_f32(arow, ks * 32, hf);

  v8f acc[8];
#pragma unroll
  for (int nt = 0; nt < 8; ++nt) acc[nt] = vzero8();

  v16bf b0 = load_frag_idx(ldsB, 0, lane);
#pragma unroll
  for (int idx = 0; idx < 32; ++idx) {
    const int ks = idx >> 3, nt = idx & 7;
    v16bf nb;
    if (idx < 31) nb = load_frag_idx(ldsB, idx + 1, lane);
    acc[nt] = wmma_bf16(afr[ks], b0, acc[nt]);
    if (idx < 31) b0 = nb;
  }

  const float e2 = 1.0f + eps2p[0];
#pragma unroll
  for (int nt = 0; nt < 8; ++nt) {
    const int col = nt * 16 + l15;
#pragma unroll
    for (int r = 0; r < 8; ++r) {
      const size_t erow = (size_t)b * 2048 + row0 + hf * 8 + r;
      e_pre[erow * DD + col] = acc[nt][r] + e2 * e_x[erow * DD + col];
    }
  }
}

// ---------------------------------------------------------------------------
// BN-MLP layer GEMM: Y = X @ W + b, plus per-column sum/sumsq for batch
// stats. Weight panel (32KB bf16, fragment-ordered) async-staged into LDS.
// ---------------------------------------------------------------------------
__global__ __launch_bounds__(256)
void gemm_bn_kernel(const float* __restrict__ X,
                    const __bf16* __restrict__ Wseg,
                    const float* __restrict__ bias,
                    float* __restrict__ Y,
                    float* __restrict__ stats) {
  __shared__ __attribute__((aligned(32))) __bf16 ldsW[16384];
  __shared__ float lsum[256];   // [0..127]=sum, [128..255]=sumsq
  const int tid = threadIdx.x;
  lsum[tid] = 0.f;
  stage_panel_async(ldsW, tid, Wseg);   // includes barrier

  const int w = tid >> 5, lane = tid & 31, hf = lane >> 4, l15 = lane & 15;
  const size_t row0 = (size_t)blockIdx.x * 128 + w * 16;
  const float* arow = X + (row0 + l15) * DD;

  v16bf afr[4];
#pragma unroll
  for (int ks = 0; ks < 4; ++ks) afr[ks] = load_a_f32(arow, ks * 32, hf);

  v8f acc[8];
#pragma unroll
  for (int nt = 0; nt < 8; ++nt) acc[nt] = vzero8();

  v16bf b0 = load_frag_idx(ldsW, 0, lane);
#pragma unroll
  for (int idx = 0; idx < 32; ++idx) {
    const int ks = idx >> 3, nt = idx & 7;
    v16bf nb;
    if (idx < 31) nb = load_frag_idx(ldsW, idx + 1, lane);
    acc[nt] = wmma_bf16(afr[ks], b0, acc[nt]);
    if (idx < 31) b0 = nb;
  }

#pragma unroll
  for (int nt = 0; nt < 8; ++nt) {
    const int col = nt * 16 + l15;
    const float bb = bias[col];
    float s = 0.f, sq = 0.f;
#pragma unroll
    for (int r = 0; r < 8; ++r) {
      const float y = acc[nt][r] + bb;
      Y[(row0 + hf * 8 + r) * DD + col] = y;
      s += y;
      sq += y * y;
    }
    s  += __shfl_xor(s, 16, 32);
    sq += __shfl_xor(sq, 16, 32);
    if (hf == 0) {
      atomicAdd(&lsum[col], s);
      atomicAdd(&lsum[128 + col], sq);
    }
  }
  __syncthreads();
  atomicAdd(&stats[tid], lsum[tid]);
}

// ---------------------------------------------------------------------------
// BatchNorm normalize + ReLU (elementwise, grid-stride).
// ---------------------------------------------------------------------------
__global__ void norm_kernel(const float* __restrict__ Y,
                            const float* __restrict__ stats,
                            const float* __restrict__ g,
                            const float* __restrict__ beta,
                            float* __restrict__ Out,
                            int Mrows) {
  const float invM = 1.0f / (float)Mrows;
  const size_t total = (size_t)Mrows * DD;
  for (size_t t = (size_t)blockIdx.x * blockDim.x + threadIdx.x; t < total;
       t += (size_t)gridDim.x * blockDim.x) {
    const int col = (int)(t & (DD - 1));
    const float mean = stats[col] * invM;
    const float var  = stats[128 + col] * invM - mean * mean;
    const float z = g[col] * (Y[t] - mean) * rsqrtf(var + BN_EPS) + beta[col];
    Out[t] = fmaxf(z, 0.f);
  }
}

// ---------------------------------------------------------------------------
extern "C" void kernel_launch(void* const* d_in, const int* in_sizes, int n_in,
                              void* d_out, int out_size, void* d_ws,
                              size_t ws_size, hipStream_t stream) {
  (void)in_sizes; (void)n_in; (void)out_size; (void)ws_size;

  const float* n_x     = (const float*)d_in[0];
  const float* e_x     = (const float*)d_in[1];
  const float* up_attr = (const float*)d_in[2];
  const float* up_adj  = (const float*)d_in[3];
  const float* badj    = (const float*)d_in[4];
  const float* W_msg   = (const float*)d_in[5];
  const float* b_msg   = (const float*)d_in[6];
  const float* W_fb    = (const float*)d_in[7];
  const float* b_fb    = (const float*)d_in[8];
  const float* eps1    = (const float*)d_in[9];
  const float* eps2    = (const float*)d_in[10];
  const float* un_W    = (const float*)d_in[11];
  const float* un_b    = (const float*)d_in[12];
  const float* un_g    = (const float*)d_in[13];
  const float* un_beta = (const float*)d_in[14];
  const float* ue_W    = (const float*)d_in[15];
  const float* ue_b    = (const float*)d_in[16];
  const float* ue_g    = (const float*)d_in[17];
  const float* ue_beta = (const float*)d_in[18];
  const int*   up_b    = (const int*)d_in[19];
  const int*   up_i    = (const int*)d_in[20];
  const int*   up_j    = (const int*)d_in[21];
  const int*   idx_i   = (const int*)d_in[22];  // up_x_i_idx
  const int*   idx_j   = (const int*)d_in[23];  // up_x_j_idx

  char* ws = (char*)d_ws;
  __bf16* wsW  = (__bf16*)ws;                    // 360 KB weight pool
  float* stats = (float*)(ws + (512 << 10));     // 6 x 256 f32 stat slots
  __bf16* nxf  = (__bf16*)(ws + (1 << 20));      // 4 MB bf16 n_x panels
  float* nA = (float*)(ws + (8 << 20));          // node ping (8 MB)
  float* nB = (float*)(ws + (16 << 20));         // node pong (8 MB)
  float* eA = (float*)(ws + (24 << 20));         // edge ping (128 MB)
  float* eB = (float*)(ws + (24 << 20) + ((size_t)ETOT * DD * 4));

  float* outN = (float*)d_out;
  float* outE = outN + (size_t)NTOT * DD;

  // Stage 0: weight swizzle + stat zeroing; n_x bf16 panels
  prep_kernel<<<dim3(48, 8), 256, 0, stream>>>(W_msg, W_fb, un_W, ue_W, wsW,
                                               stats);
  prep_nx_kernel<<<512, 256, 0, stream>>>(n_x, nxf);

  // Stage 1: fused message GEMMs + gate + weighted segment sum + residual
  msg_kernel<<<EUP / 128, 256, 0, stream>>>(
      n_x, up_attr, up_adj, b_msg, b_fb, up_b, up_i, up_j, idx_i, idx_j,
      wsW + WOFF_MSG, wsW + WOFF_FB, eps1, nA);

  // Stage 2: boundary einsum + residual
  boundary_kernel<<<NGR * 16, 256, 0, stream>>>(nxf, e_x, badj, eps2, eA);

  // Stage 3: node-branch BN-MLP (3 layers), ping-pong nA/nB, last -> d_out
  gemm_bn_kernel<<<NTOT / 128, 256, 0, stream>>>(nA, wsW + WOFF_UN,
                                                 un_b + 0, nB, stats + 0);
  norm_kernel<<<256, 256, 0, stream>>>(nB, stats + 0, un_g + 0, un_beta + 0,
                                       nB, NTOT);
  gemm_bn_kernel<<<NTOT / 128, 256, 0, stream>>>(nB, wsW + WOFF_UN + 16384,
                                                 un_b + 128, nA, stats + 256);
  norm_kernel<<<256, 256, 0, stream>>>(nA, stats + 256, un_g + 128,
                                       un_beta + 128, nA, NTOT);
  gemm_bn_kernel<<<NTOT / 128, 256, 0, stream>>>(nA, wsW + WOFF_UN + 32768,
                                                 un_b + 256, nB, stats + 512);
  norm_kernel<<<256, 256, 0, stream>>>(nB, stats + 512, un_g + 256,
                                       un_beta + 256, outN, NTOT);

  // Stage 4: edge-branch BN-MLP (3 layers), ping-pong eA/eB, last -> d_out
  gemm_bn_kernel<<<ETOT / 128, 256, 0, stream>>>(eA, wsW + WOFF_UE,
                                                 ue_b + 0, eB, stats + 768);
  norm_kernel<<<2048, 256, 0, stream>>>(eB, stats + 768, ue_g + 0, ue_beta + 0,
                                        eB, ETOT);
  gemm_bn_kernel<<<ETOT / 128, 256, 0, stream>>>(eB, wsW + WOFF_UE + 16384,
                                                 ue_b + 128, eA, stats + 1024);
  norm_kernel<<<2048, 256, 0, stream>>>(eA, stats + 1024, ue_g + 128,
                                        ue_beta + 128, eA, ETOT);
  gemm_bn_kernel<<<ETOT / 128, 256, 0, stream>>>(eA, wsW + WOFF_UE + 32768,
                                                 ue_b + 256, eB, stats + 1280);
  norm_kernel<<<2048, 256, 0, stream>>>(eB, stats + 1280, ue_g + 256,
                                        ue_beta + 256, outE, ETOT);
}
